// HeterogeneousGraphNetwork_764504178987
// MI455X (gfx1250) — matrix operand order repaired and checked by reading, not compiled, not run on hardware
//
#include <hip/hip_runtime.h>
#include <hip/hip_bf16.h>

typedef __attribute__((ext_vector_type(16))) _Float16 v16h;
typedef __attribute__((ext_vector_type(8)))  _Float16 v8h;
typedef __attribute__((ext_vector_type(8)))  float    v8f;

#define N_NODES 50000
#define N_EDGES 500000
#define HID     128
#define NHEAD   4
#define CH      32
#define NEG_SLOPE 0.2f
#define LDS_PAD 8   // halves; keeps stride K+8 (multiple of 8 -> 16B-aligned v8h rows)

// ---------------------------------------------------------------------------
// GEMM: P[M x 128] = A[M x K] @ W[K x 128] (+ bias), f16 WMMA, f32 accumulate.
// Block = 256 threads = 8 waves = one 16-row stripe x all 8 column tiles.
// W is staged ONCE per block into LDS as f16 in B-operand-major layout
// (sW[n][k], padded stride), the 16-row A tile staged as f16 row-major.
// Per K-step each wave then needs only aligned LDS vector loads:
//   A 16x32 f16 layout: lane m=lane&15; halves j<8 -> K=k0+8*hi+j,
//                       halves j>=8 -> K=k0+16+8*hi+(j-8)   (two 16B runs)
//   B 32x16 f16 layout: lane n; halves j -> same K as A     (two 16B runs)
// Grid is exact (M/16 blocks), so EXEC is all-ones at every WMMA.
// ---------------------------------------------------------------------------
__global__ void __launch_bounds__(256) gemm128_wmma(const float* __restrict__ A,
                                                    const float* __restrict__ W,
                                                    const float* __restrict__ bias,
                                                    float* __restrict__ P,
                                                    int K)
{
    __shared__ _Float16 sW[HID * (HID + LDS_PAD)];   // B operand: sW[n*S + k]
    __shared__ _Float16 sA[16  * (HID + LDS_PAD)];   // A operand: sA[m*S + k]

    const int S    = K + LDS_PAD;
    const int mt   = blockIdx.x;
    const int lane = threadIdx.x & 31;
    const int nt   = threadIdx.x >> 5;               // wave id == column tile
    const int hi   = lane >> 4;
    const int l15  = lane & 15;
    const int n    = nt * 16 + l15;                  // output column of this lane

    // stage W (row-major [k][n] f32) -> LDS f16 transposed sW[n*S + k]
    for (int idx = threadIdx.x; idx < K * HID; idx += 256) {
        int k = idx >> 7, nn = idx & 127;
        sW[nn * S + k] = (_Float16)W[idx];
    }
    // stage the 16-row A tile -> LDS f16 row-major sA[m*S + k]
    const float* Ab = A + (size_t)mt * 16 * K;
    for (int idx = threadIdx.x; idx < 16 * K; idx += 256) {
        int m = idx / K, k = idx - m * K;
        sA[m * S + k] = (_Float16)Ab[idx];
    }
    __syncthreads();

    const _Float16* arow = &sA[l15 * S];
    const _Float16* brow = &sW[n * S];

    v8f acc = {};
    for (int k0 = 0; k0 < K; k0 += 32) {
        v8h a0 = *(const v8h*)(arow + k0 + 8 * hi);        // K = k0+8hi .. +8
        v8h a1 = *(const v8h*)(arow + k0 + 16 + 8 * hi);   // K = k0+16+8hi .. +8
        v8h b0 = *(const v8h*)(brow + k0 + 8 * hi);        // K = k0+8hi .. +8
        v8h b1 = *(const v8h*)(brow + k0 + 16 + 8 * hi);   // K = k0+16+8hi .. +8
        v16h a, b;
#pragma unroll
        for (int j = 0; j < 8; ++j) {
            a[j] = a0[j]; a[8 + j] = a1[j];
            b[j] = b0[j]; b[8 + j] = b1[j];
        }
        acc = __builtin_amdgcn_wmma_f32_16x16x32_f16(
            /*neg_a=*/false, a, /*neg_b=*/false, b,
            /*c_mod=*/(short)0, acc, /*reuse_a=*/false, /*reuse_b=*/false);
    }

    const float bn = bias ? bias[n] : 0.0f;
#pragma unroll
    for (int r = 0; r < 8; ++r) {
        int mm = mt * 16 + r + 8 * hi;                     // D: VGPR r -> row r + 8*hi
        P[(size_t)mm * HID + n] = acc[r] + bn;
    }
}

// ---------------------------------------------------------------------------
// Per-node attention scores: s_src/s_dst[n,h] = dot(xp[n, h*32 : h*32+32], att)
// ---------------------------------------------------------------------------
__global__ void gat_scores(const float* __restrict__ xp,
                           const float* __restrict__ att_src,
                           const float* __restrict__ att_dst,
                           float* __restrict__ s_src,
                           float* __restrict__ s_dst)
{
    int idx = blockIdx.x * blockDim.x + threadIdx.x;
    if (idx >= N_NODES * NHEAD) return;
    int nd = idx >> 2, h = idx & 3;
    const float* row = xp + (size_t)nd * HID + h * CH;
    const float* as  = att_src + h * CH;
    const float* ad  = att_dst + h * CH;
    float ss = 0.f, sd = 0.f;
#pragma unroll
    for (int c = 0; c < CH; ++c) { float v = row[c]; ss += v * as[c]; sd += v * ad[c]; }
    s_src[idx] = ss;
    s_dst[idx] = sd;
}

__global__ void init_softmax(float* __restrict__ mmax, float* __restrict__ denom)
{
    int idx = blockIdx.x * blockDim.x + threadIdx.x;
    if (idx >= N_NODES * NHEAD) return;
    mmax[idx]  = -1e30f;
    denom[idx] = 0.0f;
}

// final = emb + bg_owns[c] + bg_shares[c]   (GAT outputs scatter-accumulate on top)
__global__ void init_final(const float* __restrict__ emb,
                           const float* __restrict__ bg0,
                           const float* __restrict__ bg1,
                           float* __restrict__ fin)
{
    int idx = blockIdx.x * blockDim.x + threadIdx.x;
    if (idx >= 3 * N_NODES * HID) return;
    int c = idx & (HID - 1);
    fin[idx] = emb[idx] + bg0[c] + bg1[c];
}

__device__ __forceinline__ float leaky(float x) { return x > 0.f ? x : NEG_SLOPE * x; }

// sign-split atomic float max (valid across mixed-sign updates; init = -1e30)
__device__ __forceinline__ void atomicMaxFloat(float* addr, float val)
{
    if (val >= 0.f) atomicMax((int*)addr, __float_as_int(val));
    else            atomicMin((unsigned int*)addr, (unsigned int)__float_as_int(val));
}

__device__ __forceinline__ void edge_sd(int i, const int* src, const int* dst, int& s, int& d)
{
    if (i < N_EDGES) { s = src[i]; d = dst[i]; }
    else             { s = d = i - N_EDGES; }      // appended self-loops
}

// Pass 1: segment max over incoming edges (per head)
__global__ void edge_max(const int* __restrict__ src, const int* __restrict__ dst,
                         const float* __restrict__ s_src, const float* __restrict__ s_dst,
                         float* __restrict__ mmax)
{
    int i = blockIdx.x * blockDim.x + threadIdx.x;
    if (i >= N_EDGES + N_NODES) return;
    int s, d; edge_sd(i, src, dst, s, d);
#pragma unroll
    for (int h = 0; h < NHEAD; ++h) {
        float e = leaky(s_src[s * NHEAD + h] + s_dst[d * NHEAD + h]);
        atomicMaxFloat(&mmax[d * NHEAD + h], e);
    }
}

// Pass 2: denom[d,h] += exp(e - m[d,h])
__global__ void edge_expsum(const int* __restrict__ src, const int* __restrict__ dst,
                            const float* __restrict__ s_src, const float* __restrict__ s_dst,
                            const float* __restrict__ mmax, float* __restrict__ denom)
{
    int i = blockIdx.x * blockDim.x + threadIdx.x;
    if (i >= N_EDGES + N_NODES) return;
    int s, d; edge_sd(i, src, dst, s, d);
#pragma unroll
    for (int h = 0; h < NHEAD; ++h) {
        float e = leaky(s_src[s * NHEAD + h] + s_dst[d * NHEAD + h]);
        atomicAdd(&denom[d * NHEAD + h], expf(e - mmax[d * NHEAD + h]));
    }
}

// Pass 3: wave-per-edge weighted scatter: fin[d, h*32+lane] += alpha_h * xp[s, h*32+lane]
__global__ void edge_scatter(const int* __restrict__ src, const int* __restrict__ dst,
                             const float* __restrict__ s_src, const float* __restrict__ s_dst,
                             const float* __restrict__ mmax, const float* __restrict__ denom,
                             const float* __restrict__ xp, float* __restrict__ fin)
{
    int lane = threadIdx.x & 31;
    int wave = threadIdx.x >> 5;
    int edge = blockIdx.x * (blockDim.x >> 5) + wave;
    if (edge >= N_EDGES + N_NODES) return;
    int s, d; edge_sd(edge, src, dst, s, d);

    float alpha = 0.f;
    if (lane < NHEAD) {
        int h = lane;
        float e = leaky(s_src[s * NHEAD + h] + s_dst[d * NHEAD + h]);
        alpha = expf(e - mmax[d * NHEAD + h]) / denom[d * NHEAD + h];
    }
    const float* xrow = xp + (size_t)s * HID;
    float* frow = fin + (size_t)d * HID;
#pragma unroll
    for (int h = 0; h < NHEAD; ++h) {
        float ah = __shfl(alpha, h, 32);
        atomicAdd(&frow[h * CH + lane], ah * xrow[h * CH + lane]);
    }
}

// wave-per-row classifier: sigmoid(dot(final[row], Wc) + bc)
__global__ void classifier(const float* __restrict__ fin,
                           const float* __restrict__ Wc,
                           const float* __restrict__ bc,
                           float* __restrict__ out)
{
    int lane = threadIdx.x & 31;
    int wave = threadIdx.x >> 5;
    int row  = blockIdx.x * (blockDim.x >> 5) + wave;
    if (row >= 3 * N_NODES) return;
    const float* r = fin + (size_t)row * HID;
    float sum = 0.f;
#pragma unroll
    for (int q = 0; q < 4; ++q) sum += r[q * 32 + lane] * Wc[q * 32 + lane];
#pragma unroll
    for (int off = 16; off > 0; off >>= 1) sum += __shfl_xor(sum, off, 32);
    if (lane == 0) out[row] = 1.f / (1.f + expf(-(sum + bc[0])));
}

// ---------------------------------------------------------------------------
extern "C" void kernel_launch(void* const* d_in, const int* in_sizes, int n_in,
                              void* d_out, int out_size, void* d_ws, size_t ws_size,
                              hipStream_t stream)
{
    const float* x_acc = (const float*)d_in[0];
    const float* x_cus = (const float*)d_in[1];
    const float* x_txn = (const float*)d_in[2];
    // d_in[3] = ei_transacts: dead code in the reference
    const int*   ei_owns = (const int*)d_in[4];
    const int*   ei_shr  = (const int*)d_in[5];
    const float* W_acc = (const float*)d_in[6];  const float* b_acc = (const float*)d_in[7];
    const float* W_cus = (const float*)d_in[8];  const float* b_cus = (const float*)d_in[9];
    const float* W_txn = (const float*)d_in[10]; const float* b_txn = (const float*)d_in[11];
    const float* Wg[2]  = {(const float*)d_in[12], (const float*)d_in[16]};
    const float* asA[2] = {(const float*)d_in[13], (const float*)d_in[17]};
    const float* adA[2] = {(const float*)d_in[14], (const float*)d_in[18]};
    const float* bg[2]  = {(const float*)d_in[15], (const float*)d_in[19]};
    const float* Wcls = (const float*)d_in[20]; const float* bcls = (const float*)d_in[21];
    float* out = (float*)d_out;

    // workspace layout (floats): ~182 MB total
    float* ws   = (float*)d_ws;
    float* emb  = ws;                                     // 3*N*128
    float* fin  = emb  + (size_t)3 * N_NODES * HID;       // 3*N*128
    float* xp   = fin  + (size_t)3 * N_NODES * HID;       // N*128
    float* ssrc = xp   + (size_t)N_NODES * HID;           // N*4
    float* sdst = ssrc + (size_t)N_NODES * NHEAD;         // N*4
    float* mmax = sdst + (size_t)N_NODES * NHEAD;         // N*4
    float* den  = mmax + (size_t)N_NODES * NHEAD;         // N*4

    const int gemmBlocks = N_NODES / 16;                  // 3125, exact
    const int nhTot = N_NODES * NHEAD;
    const int nhBlk = (nhTot + 255) / 256;
    const int ET    = N_EDGES + N_NODES;

    // per-type embeddings
    gemm128_wmma<<<gemmBlocks, 256, 0, stream>>>(x_acc, W_acc, b_acc,
        emb + (size_t)0 * N_NODES * HID, 64);
    gemm128_wmma<<<gemmBlocks, 256, 0, stream>>>(x_cus, W_cus, b_cus,
        emb + (size_t)1 * N_NODES * HID, 32);
    gemm128_wmma<<<gemmBlocks, 256, 0, stream>>>(x_txn, W_txn, b_txn,
        emb + (size_t)2 * N_NODES * HID, 128);

    { int tot = 3 * N_NODES * HID;
      init_final<<<(tot + 255) / 256, 256, 0, stream>>>(emb, bg[0], bg[1], fin); }

    for (int t = 0; t < 3; ++t) {
        float* embt = emb + (size_t)t * N_NODES * HID;
        float* fint = fin + (size_t)t * N_NODES * HID;
        for (int g = 0; g < 2; ++g) {
            const int* src = (g == 0) ? ei_owns : ei_shr;
            const int* dst = src + N_EDGES;
            gemm128_wmma<<<gemmBlocks, 256, 0, stream>>>(embt, Wg[g], nullptr, xp, 128);
            gat_scores  <<<nhBlk, 256, 0, stream>>>(xp, asA[g], adA[g], ssrc, sdst);
            init_softmax<<<nhBlk, 256, 0, stream>>>(mmax, den);
            edge_max    <<<(ET + 255) / 256, 256, 0, stream>>>(src, dst, ssrc, sdst, mmax);
            edge_expsum <<<(ET + 255) / 256, 256, 0, stream>>>(src, dst, ssrc, sdst, mmax, den);
            edge_scatter<<<(ET + 7) / 8, 256, 0, stream>>>(src, dst, ssrc, sdst, mmax, den, xp, fint);
        }
    }

    classifier<<<(3 * N_NODES + 7) / 8, 256, 0, stream>>>(fin, Wcls, bcls, out);
}
